// MathMirror_21337397526889
// MI455X (gfx1250) — compile-verified, matrix operand order checked by reading, not run on hardware
//
#include <hip/hip_runtime.h>
#include <hip/hip_bf16.h>
#include <math.h>

// ---------------- problem constants ----------------
#define Bc  4
#define Tc  2048
#define Dc  256
#define Hc  8
#define HDc 32
#define Lc  12
#define Vc  256
#define MTc (Bc * Tc)          // 8192 token rows

typedef __attribute__((ext_vector_type(16))) __bf16 v16bf;
typedef __attribute__((ext_vector_type(8)))  float  v8f;

// f32 -> bf16 round-to-nearest-even
__device__ inline __bf16 f2bf(float f) {
  unsigned u = __builtin_bit_cast(unsigned, f);
  unsigned r = (u + 0x7FFFu + ((u >> 16) & 1u)) >> 16;
  unsigned short s = (unsigned short)r;
  return __builtin_bit_cast(__bf16, s);
}

struct U8x { uint4 a, b; };

// One lane's 16-element bf16 fragment slice from a row-major K-run.
// CDNA5 16-bit A/B layout: elems 0..7 = K kb..kb+7, elems 8..15 = K 16+kb..16+kb+7,
// kb = (lane>=16) ? 8 : 0.  Both 16B chunks are aligned for all our buffers.
__device__ inline v16bf load_frag_row(const __bf16* rowbase, int kb) {
  U8x u;
  u.a = *reinterpret_cast<const uint4*>(reinterpret_cast<const char*>(rowbase) + kb * 2);
  u.b = *reinterpret_cast<const uint4*>(reinterpret_cast<const char*>(rowbase) + (16 + kb) * 2);
  return __builtin_bit_cast(v16bf, u);
}

__device__ inline v8f wmma_bf16(v16bf a, v16bf b, v8f c) {
  return __builtin_amdgcn_wmma_f32_16x16x32_bf16(false, a, false, b, (short)0, c, false, false);
}

// Async copy of 32 bytes global -> LDS per lane (two b128 transfers),
// tracked by ASYNCcnt; completes without touching VGPRs.
__device__ inline void async_copy_32B(__bf16* lds_dst, const __bf16* gsrc) {
  unsigned l0 = (unsigned)(unsigned long)lds_dst;
  asm volatile(
      "global_load_async_to_lds_b128 %0, %2, off\n\t"
      "global_load_async_to_lds_b128 %1, %2, off offset:16"
      :
      : "v"(l0), "v"(l0 + 16u), "v"(gsrc)
      : "memory");
}

__device__ inline void wait_async0() {
  asm volatile("s_wait_asynccnt 0" ::: "memory");
}

// ---------------- one-time weight transform: f32 [L,K,N] -> bf16 [L,N,K] ----------------
__global__ __launch_bounds__(256) void cvt_transpose_kernel(const float* __restrict__ in,
                                                            __bf16* __restrict__ out,
                                                            int K, int N, long total) {
  long i = (long)blockIdx.x * 256 + threadIdx.x;
  if (i >= total) return;
  long kn = (long)K * N;
  long l = i / kn;
  long r = i - l * kn;
  int k = (int)(r / N);
  int n = (int)(r - (long)k * N);
  out[l * kn + (long)n * K + k] = f2bf(in[i]);
}

// ---------------- simple kernels ----------------
__global__ __launch_bounds__(256) void embed_kernel(const int* __restrict__ x,
                                                    const float* __restrict__ tok,
                                                    const float* __restrict__ pos,
                                                    float* __restrict__ h) {
  long i = (long)blockIdx.x * 256 + threadIdx.x;   // over MT*D
  int d = (int)(i & (Dc - 1));
  long bt = i >> 8;
  int t = (int)(bt & (Tc - 1));
  int tokid = x[bt];
  h[i] = tok[(long)tokid * Dc + d] + pos[(long)t * Dc + d];
}

// LayerNorm over D=256, one row per block; writes bf16 (GEMM A input)
__global__ __launch_bounds__(256) void ln_kernel(const float* __restrict__ x,
                                                 const float* __restrict__ w,
                                                 const float* __restrict__ b,
                                                 __bf16* __restrict__ out) {
  __shared__ float red[256];
  int row = blockIdx.x;
  int t = threadIdx.x;
  float xv = x[(long)row * Dc + t];
  red[t] = xv;
  __syncthreads();
  for (int s = 128; s > 0; s >>= 1) { if (t < s) red[t] += red[t + s]; __syncthreads(); }
  float mean = red[0] * (1.0f / Dc);
  __syncthreads();
  float d = xv - mean;
  red[t] = d * d;
  __syncthreads();
  for (int s = 128; s > 0; s >>= 1) { if (t < s) red[t] += red[t + s]; __syncthreads(); }
  float var = red[0] * (1.0f / Dc);
  float y = d * rsqrtf(var + 1e-5f) * w[t] + b[t];
  out[(long)row * Dc + t] = f2bf(y);
}

// qkv f32 [B,T,3,H,HD] -> q,k bf16 [B,H,T,HD]; v bf16 TRANSPOSED [B,H,HD,T]
__global__ __launch_bounds__(256) void split_qkv_kernel(const float* __restrict__ qkv,
                                                        __bf16* __restrict__ q,
                                                        __bf16* __restrict__ k,
                                                        __bf16* __restrict__ v) {
  long i = (long)blockIdx.x * 256 + threadIdx.x;   // over MT*3*D
  int dd = (int)(i & (HDc - 1));
  long r = i >> 5;
  int hh = (int)(r & (Hc - 1)); r >>= 3;
  int c  = (int)(r % 3);        r /= 3;
  int t  = (int)(r & (Tc - 1));
  int b  = (int)(r >> 11);
  __bf16 val = f2bf(qkv[i]);
  if (c == 0) {
    q[(((long)(b * Hc + hh) * Tc) + t) * HDc + dd] = val;
  } else if (c == 1) {
    k[(((long)(b * Hc + hh) * Tc) + t) * HDc + dd] = val;
  } else {
    v[(((long)(b * Hc + hh) * HDc) + dd) * Tc + t] = val;   // transposed
  }
}

// ---------------- WMMA GEMM ----------------
// C[M,N] = A[M,K](bf16,row-major) @ W^T (bf16 stored [N,K]) + bias.
// Block tile 128x64, 8 waves; each wave owns a 32x32 macro-tile = 4 WMMA accumulators.
// A staged in double-buffered LDS via GLOBAL_LOAD_ASYNC_TO_LDS_B128 (ASYNCcnt);
// the two buffers are processed with STATIC indices (2 K-steps per loop trip,
// K % 64 == 0 here) so there is no dynamic buffer state in the hot loop.
// B fragments read straight from L2-resident transposed weights (b128 loads).
// mode: 0 = f32 out, 1 = f32 out + residual, 2 = exact GELU then bf16 out.
__global__ __launch_bounds__(256) void gemm_bf16_kernel(
    const __bf16* __restrict__ A, const __bf16* __restrict__ Wt,
    const float* __restrict__ bias, const float* __restrict__ res,
    float* __restrict__ outf, __bf16* __restrict__ outb,
    int M, int N, int K, int mode) {
  __shared__ __bf16 As[2][128][40];
  int tid  = threadIdx.x;
  int lane = tid & 31;
  int wave = tid >> 5;
  int wm = wave >> 1;            // 0..3 : 32-row slice of 128
  int wn = wave & 1;             // 0..1 : 32-col slice of 64
  int m0 = blockIdx.x * 128;
  int n0 = blockIdx.y * 64;

  int ml = lane & 15;
  int kb = (lane >> 4) * 8;

  int lr = tid >> 1;             // 0..127 : A row staged by this thread
  int lc = (tid & 1) * 16;       // 0 or 16

  const __bf16* arow  = A + (long)(m0 + lr) * K + lc;
  const __bf16* wrow0 = Wt + (long)(n0 + wn * 32 + ml) * K;
  const __bf16* wrow1 = Wt + (long)(n0 + wn * 32 + 16 + ml) * K;

  const __bf16* a0p = &As[0][wm * 32 + ml][0];
  const __bf16* a1p = &As[0][wm * 32 + 16 + ml][0];
  const __bf16* b0p = &As[1][wm * 32 + ml][0];
  const __bf16* b1p = &As[1][wm * 32 + 16 + ml][0];

  // preload tile 0 into buffer 0
  async_copy_32B(&As[0][lr][lc], arow);

  v8f acc00 = {}, acc01 = {}, acc10 = {}, acc11 = {};
  for (int k0 = 0; k0 < K; k0 += 64) {
    // ---- half 1: compute buffer 0 (tile k0), stream tile k0+32 into buffer 1
    wait_async0();
    __syncthreads();
    async_copy_32B(&As[1][lr][lc], arow + k0 + 32);
    {
      v16bf af0 = load_frag_row(a0p, kb);
      v16bf af1 = load_frag_row(a1p, kb);
      v16bf bf0 = load_frag_row(wrow0 + k0, kb);
      v16bf bf1 = load_frag_row(wrow1 + k0, kb);
      acc00 = wmma_bf16(af0, bf0, acc00);
      acc01 = wmma_bf16(af0, bf1, acc01);
      acc10 = wmma_bf16(af1, bf0, acc10);
      acc11 = wmma_bf16(af1, bf1, acc11);
    }
    // ---- half 2: compute buffer 1 (tile k0+32), stream tile k0+64 into buffer 0
    wait_async0();
    __syncthreads();
    if (k0 + 64 < K)
      async_copy_32B(&As[0][lr][lc], arow + k0 + 64);
    {
      v16bf af0 = load_frag_row(b0p, kb);
      v16bf af1 = load_frag_row(b1p, kb);
      v16bf bf0 = load_frag_row(wrow0 + k0 + 32, kb);
      v16bf bf1 = load_frag_row(wrow1 + k0 + 32, kb);
      acc00 = wmma_bf16(af0, bf0, acc00);
      acc01 = wmma_bf16(af0, bf1, acc01);
      acc10 = wmma_bf16(af1, bf0, acc10);
      acc11 = wmma_bf16(af1, bf1, acc11);
    }
  }

  int half = lane >> 4;
  int colA = n0 + wn * 32 + ml;
  int colB = colA + 16;
  float bvA = bias[colA];
  float bvB = bias[colB];
  #pragma unroll
  for (int mt = 0; mt < 2; ++mt) {
    v8f aN = mt ? acc10 : acc00;
    v8f aB = mt ? acc11 : acc01;
    #pragma unroll
    for (int vv = 0; vv < 8; ++vv) {
      int row = m0 + wm * 32 + mt * 16 + vv + half * 8;
      long ia = (long)row * N + colA;
      long ib = (long)row * N + colB;
      float va = aN[vv] + bvA;
      float vb = aB[vv] + bvB;
      if (mode == 0) {
        outf[ia] = va;
        outf[ib] = vb;
      } else if (mode == 1) {
        outf[ia] = res[ia] + va;
        outf[ib] = res[ib] + vb;
      } else {
        float ga = 0.5f * va * (1.0f + erff(va * 0.70710678118654752f));
        float gb = 0.5f * vb * (1.0f + erff(vb * 0.70710678118654752f));
        outb[ia] = f2bf(ga);
        outb[ib] = f2bf(gb);
      }
    }
  }
}

// ---------------- flash attention ----------------
// One wave per 16-row q tile; score WMMA K == HD == 32.  Online softmax in the
// WMMA C layout; P moved C->A layout through per-wave LDS; V read from the
// transposed [B,H,HD,T] buffer so every fragment is 2 contiguous b128 loads.
__global__ __launch_bounds__(256) void attn_kernel(const __bf16* __restrict__ q,
                                                   const __bf16* __restrict__ k,
                                                   const __bf16* __restrict__ vT,
                                                   __bf16* __restrict__ o) {
  __shared__ __bf16 Pl[8][16][32];
  int lane = threadIdx.x & 31;
  int w = threadIdx.x >> 5;
  int gw = blockIdx.x * 8 + w;     // B*H*(T/16) = 4096 q tiles
  int qt = gw & 127;               // T/16 = 128
  int bh = gw >> 7;
  int hh = bh & (Hc - 1);
  int b  = bh >> 3;

  const __bf16* qp  = q  + ((long)(b * Hc + hh) * Tc) * HDc;
  const __bf16* kp  = k  + ((long)(b * Hc + hh) * Tc) * HDc;
  const __bf16* vTp = vT + ((long)(b * Hc + hh) * HDc) * Tc;

  int ml = lane & 15;
  int half = lane >> 4;
  int kb = half * 8;

  v16bf qf = load_frag_row(qp + (qt * 16 + ml) * HDc, kb);
  const __bf16* vrow0 = vTp + (long)ml * Tc;          // hd = ml
  const __bf16* vrow1 = vTp + (long)(16 + ml) * Tc;   // hd = 16+ml

  float mrow[8], lrow[8];
  v8f O0 = {}, O1 = {};
  #pragma unroll
  for (int i = 0; i < 8; ++i) { mrow[i] = -3.0e38f; lrow[i] = 0.0f; }

  const float scale = 0.17677669529663687f;   // 1/sqrt(HD)
  int qend = qt * 16 + 15;
  for (int jj = 0; jj <= qend; jj += 32) {
    v16bf kf0 = load_frag_row(kp + (jj + ml) * HDc, kb);
    v16bf kf1 = load_frag_row(kp + (jj + 16 + ml) * HDc, kb);
    v8f z = {};
    v8f s0 = wmma_bf16(qf, kf0, z);
    v8f s1 = wmma_bf16(qf, kf1, z);

    int c0 = jj + ml, c1 = jj + 16 + ml;
    #pragma unroll
    for (int vv = 0; vv < 8; ++vv) {
      int row = qt * 16 + vv + half * 8;
      float a0 = s0[vv] * scale; if (c0 > row) a0 = -1.0e30f;
      float a1 = s1[vv] * scale; if (c1 > row) a1 = -1.0e30f;
      float t = fmaxf(a0, a1);
      #pragma unroll
      for (int msk = 1; msk < 16; msk <<= 1) t = fmaxf(t, __shfl_xor(t, msk, 32));
      float mn = fmaxf(mrow[vv], t);
      float ef = __expf(mrow[vv] - mn);
      mrow[vv] = mn;
      float p0 = __expf(a0 - mn);
      float p1 = __expf(a1 - mn);
      float ps = p0 + p1;
      #pragma unroll
      for (int msk = 1; msk < 16; msk <<= 1) ps += __shfl_xor(ps, msk, 32);
      lrow[vv] = lrow[vv] * ef + ps;
      O0[vv] *= ef;
      O1[vv] *= ef;
      int r = vv + half * 8;
      Pl[w][r][ml]      = f2bf(p0);
      Pl[w][r][16 + ml] = f2bf(p1);
    }
    __builtin_amdgcn_wave_barrier();
    v16bf pf = load_frag_row(&Pl[w][ml][0], kb);   // A-layout read of P
    __builtin_amdgcn_wave_barrier();
    v16bf vf0 = load_frag_row(vrow0 + jj, kb);     // B operand: V^T rows
    v16bf vf1 = load_frag_row(vrow1 + jj, kb);
    O0 = wmma_bf16(pf, vf0, O0);
    O1 = wmma_bf16(pf, vf1, O1);
  }

  #pragma unroll
  for (int vv = 0; vv < 8; ++vv) {
    float inv = 1.0f / lrow[vv];
    int row = qt * 16 + vv + half * 8;
    __bf16* op = o + ((long)(b * Tc + row)) * Dc + hh * HDc;
    op[ml]      = f2bf(O0[vv] * inv);
    op[16 + ml] = f2bf(O1[vv] * inv);
  }
}

// ---------------- host orchestration ----------------
extern "C" void kernel_launch(void* const* d_in, const int* in_sizes, int n_in,
                              void* d_out, int out_size, void* d_ws, size_t ws_size,
                              hipStream_t stream) {
  (void)in_sizes; (void)n_in; (void)out_size; (void)ws_size;
  const int*   x      = (const int*)d_in[0];
  const float* tok    = (const float*)d_in[1];
  const float* pos    = (const float*)d_in[2];
  const float* ln1_w  = (const float*)d_in[3];
  const float* ln1_b  = (const float*)d_in[4];
  const float* qkv_w  = (const float*)d_in[5];
  const float* qkv_b  = (const float*)d_in[6];
  const float* out_w  = (const float*)d_in[7];
  const float* out_b  = (const float*)d_in[8];
  const float* ln2_w  = (const float*)d_in[9];
  const float* ln2_b  = (const float*)d_in[10];
  const float* fc1_w  = (const float*)d_in[11];
  const float* fc1_b  = (const float*)d_in[12];
  const float* fc2_w  = (const float*)d_in[13];
  const float* fc2_b  = (const float*)d_in[14];
  const float* lnf_w  = (const float*)d_in[15];
  const float* lnf_b  = (const float*)d_in[16];
  const float* head_w = (const float*)d_in[17];
  const float* head_b = (const float*)d_in[18];

  char* ws = (char*)d_ws;
  size_t off = 0;
  auto alloc = [&](size_t bytes) -> void* {
    void* p = ws + off;
    off += (bytes + 255) & ~(size_t)255;
    return p;
  };
  const long MT = MTc;
  float*  h     = (float*)alloc(MT * Dc * 4);
  __bf16* z     = (__bf16*)alloc(MT * Dc * 2);
  float*  qkvf  = (float*)alloc(MT * 3 * Dc * 4);
  __bf16* qbf   = (__bf16*)alloc(MT * Dc * 2);
  __bf16* kbf   = (__bf16*)alloc(MT * Dc * 2);
  __bf16* vbf   = (__bf16*)alloc(MT * Dc * 2);       // transposed [B,H,HD,T]
  __bf16* obf   = (__bf16*)alloc(MT * Dc * 2);
  __bf16* g1bf  = (__bf16*)alloc(MT * 4 * Dc * 2);
  __bf16* wqkv  = (__bf16*)alloc((long)Lc * Dc * 3 * Dc * 2);   // [L, 3D, D]
  __bf16* wout  = (__bf16*)alloc((long)Lc * Dc * Dc * 2);       // [L, D, D]
  __bf16* wfc1  = (__bf16*)alloc((long)Lc * Dc * 4 * Dc * 2);   // [L, 4D, D]
  __bf16* wfc2  = (__bf16*)alloc((long)Lc * 4 * Dc * Dc * 2);   // [L, D, 4D]
  __bf16* whead = (__bf16*)alloc((long)Dc * Vc * 2);            // [V, D]

  dim3 blk(256);
  auto cvtT = [&](const float* src, __bf16* dst, int K, int N, long total) {
    cvt_transpose_kernel<<<dim3((unsigned)((total + 255) / 256)), blk, 0, stream>>>(
        src, dst, K, N, total);
  };
  cvtT(qkv_w,  wqkv,  Dc, 3 * Dc, (long)Lc * Dc * 3 * Dc);
  cvtT(out_w,  wout,  Dc, Dc,     (long)Lc * Dc * Dc);
  cvtT(fc1_w,  wfc1,  Dc, 4 * Dc, (long)Lc * Dc * 4 * Dc);
  cvtT(fc2_w,  wfc2,  4 * Dc, Dc, (long)Lc * 4 * Dc * Dc);
  cvtT(head_w, whead, Dc, Vc,     (long)Dc * Vc);

  embed_kernel<<<dim3((unsigned)(MT * Dc / 256)), blk, 0, stream>>>(x, tok, pos, h);

  auto gemm = [&](const __bf16* A, const __bf16* Wt, const float* bias,
                  const float* res, float* outf, __bf16* outb,
                  int M, int N, int K, int mode) {
    dim3 grid(M / 128, N / 64);
    gemm_bf16_kernel<<<grid, blk, 0, stream>>>(A, Wt, bias, res, outf, outb, M, N, K, mode);
  };

  for (int l = 0; l < Lc; ++l) {
    ln_kernel<<<dim3((unsigned)MT), blk, 0, stream>>>(h, ln1_w + l * Dc, ln1_b + l * Dc, z);
    gemm(z, wqkv + (long)l * Dc * 3 * Dc, qkv_b + l * 3 * Dc,
         nullptr, qkvf, nullptr, MTc, 3 * Dc, Dc, 0);
    split_qkv_kernel<<<dim3((unsigned)(MT * 3 * Dc / 256)), blk, 0, stream>>>(qkvf, qbf, kbf, vbf);
    attn_kernel<<<dim3((unsigned)(Bc * Hc * (Tc / 16) / 8)), blk, 0, stream>>>(qbf, kbf, vbf, obf);
    gemm(obf, wout + (long)l * Dc * Dc, out_b + l * Dc,
         h, h, nullptr, MTc, Dc, Dc, 1);
    ln_kernel<<<dim3((unsigned)MT), blk, 0, stream>>>(h, ln2_w + l * Dc, ln2_b + l * Dc, z);
    gemm(z, wfc1 + (long)l * Dc * 4 * Dc, fc1_b + l * 4 * Dc,
         nullptr, nullptr, g1bf, MTc, 4 * Dc, Dc, 2);
    gemm(g1bf, wfc2 + (long)l * 4 * Dc * Dc, fc2_b + l * Dc,
         h, h, nullptr, MTc, Dc, 4 * Dc, 1);
  }

  ln_kernel<<<dim3((unsigned)MT), blk, 0, stream>>>(h, lnf_w, lnf_b, z);
  gemm(z, whead, head_b, nullptr, (float*)d_out, nullptr, MTc, Vc, Dc, 0);
}
